// StateSlotBank_16793322127857
// MI455X (gfx1250) — matrix operand order repaired and checked by reading, not compiled, not run on hardware
//
#include <hip/hip_runtime.h>
#include <stdint.h>

#define BB   8
#define SS   128
#define DIN  768
#define NS   64
#define DD   128
#define NH   4
#define DHD  32
#define MAXT 8
#define THRV 0.5f
#define EPSV 1e-5f
#define NT   256

typedef __attribute__((ext_vector_type(16))) __bf16 bf16x16;
typedef __attribute__((ext_vector_type(8)))  __bf16 bf16x8;
typedef __attribute__((ext_vector_type(2)))  __bf16 bf16x2;
typedef __attribute__((ext_vector_type(8)))  float  f32x8;

__device__ __forceinline__ float sigf(float x) { return 1.0f / (1.0f + __expf(-x)); }

// gfx1250 async global->LDS copy (ASYNCcnt-tracked), per CDNA5 ISA
__device__ __forceinline__ void async_cp16(void* lds, const void* g) {
  asm volatile("global_load_async_to_lds_b128 %0, %1, off"
               :: "v"((uint32_t)(uintptr_t)(lds)), "v"((uint64_t)(uintptr_t)(g))
               : "memory");
}
__device__ __forceinline__ void async_wait0() {
  asm volatile("s_wait_asynccnt 0" ::: "memory");
}

__global__ __launch_bounds__(NT) void slotbank_kernel(
    const float* __restrict__ x,       const float* __restrict__ slot_mem,
    const float* __restrict__ ln_w,    const float* __restrict__ ln_b,
    const float* __restrict__ w_in,    const float* __restrict__ w_q,
    const float* __restrict__ w_k,     const float* __restrict__ w_v,
    const float* __restrict__ w_gate,  const float* __restrict__ b_gate,
    const float* __restrict__ halt_w1, const float* __restrict__ halt_b1,
    const float* __restrict__ halt_w2, const float* __restrict__ halt_b2,
    const float* __restrict__ w_out,   const float* __restrict__ dn_wk,
    const float* __restrict__ dn_wv,   const float* __restrict__ dn_wb,
    const float* __restrict__ dn_bb,   const float* __restrict__ dn_lam,
    const float* __restrict__ dn_wo,   float* __restrict__ out,
    float* __restrict__ ws_sum, unsigned* __restrict__ ws_cnt)
{
  __shared__ float  s_slots[NS * DD];     // persistent slot state (fp32)
  __shared__ __bf16 s_abf[NS * DD];       // h_new staged as bf16 WMMA A
  __shared__ float  s_x[2][DIN];          // async double-buffered x rows
  __shared__ float  s_xp[DD], s_q[DD], s_wkq[DD];
  __shared__ float  s_part[NT];
  __shared__ float  s_attn[NS], s_red[NS];
  __shared__ float  s_asl[DD], s_read[DD];
  __shared__ float  s_h1[NS], s_gate[NS];
  __shared__ float  s_kv[DD], s_vv[DD];
  __shared__ float  s_beta[NH], s_nrm[NH];
  __shared__ float  s_kh[NS * NH];
  __shared__ float  s_acc[DD], s_lam[DD];
  __shared__ float  s_scal[4];            // [0]=halt [1]=rem [2]=mean(halt)

  const int tid  = threadIdx.x;
  const int lane = tid & 31;              // wave32
  const int wv   = tid >> 5;              // 8 waves
  const int b    = blockIdx.x;

  // ---- persistent state init (vectorized b128) ----
  #pragma unroll
  for (int i = 0; i < (NS * DD) / (4 * NT); ++i)
    ((float4*)s_slots)[tid + i * NT] = ((const float4*)slot_mem)[tid + i * NT];
  if (tid < DD) s_lam[tid] = dn_lam[tid];

  // ---- pin dn_wo B-fragments (32x16 bf16 per k-tile) in VGPRs for whole kernel ----
  // lanes 0-15: col=lane,  comps c <-> K=c ; lanes 16-31: col=lane-16, comps c <-> K=16+c
  const int colB  = wv * 16 + (lane & 15);
  const int kbase = (lane >> 4) * 16;
  bf16x16 bfr[4];
  #pragma unroll
  for (int kt = 0; kt < 4; ++kt) {
    bf16x16 tmp = {};
    #pragma unroll
    for (int j = 0; j < 16; ++j)
      tmp[j] = (__bf16)dn_wo[(kt * 32 + kbase + j) * DD + colB];
    bfr[kt] = tmp;
  }

  // stage x row 0 asynchronously (192 lanes x 16B = 3 KB)
  if (tid < DIN / 4)
    async_cp16(&s_x[0][tid * 4], x + ((size_t)b * SS + 0) * DIN + tid * 4);
  __syncthreads();

  const float rscale = 0.08838834764831845f; // 1/sqrt(128)

  for (int s = 0; s < SS; ++s) {
    float* xr = s_x[s & 1];
    async_wait0();
    __syncthreads();
    if (s + 1 < SS && tid < DIN / 4)   // prefetch next row into other buffer
      async_cp16(&s_x[(s + 1) & 1][tid * 4],
                 x + ((size_t)b * SS + (s + 1)) * DIN + tid * 4);

    if (tid < DD) s_acc[tid] = 0.f;
    if (tid == 0) s_scal[1] = 1.f;

    // ---- LayerNorm over 768, in place ----
    float p0 = xr[tid], p1 = xr[tid + NT], p2 = xr[tid + 2 * NT];
    s_part[tid] = p0 + p1 + p2;
    __syncthreads();
    for (int off = NT / 2; off > 0; off >>= 1) {
      if (tid < off) s_part[tid] += s_part[tid + off];
      __syncthreads();
    }
    float mu = s_part[0] * (1.f / DIN);
    __syncthreads();
    float d0 = p0 - mu, d1 = p1 - mu, d2 = p2 - mu;
    s_part[tid] = d0 * d0 + d1 * d1 + d2 * d2;
    __syncthreads();
    for (int off = NT / 2; off > 0; off >>= 1) {
      if (tid < off) s_part[tid] += s_part[tid + off];
      __syncthreads();
    }
    float rstd = rsqrtf(s_part[0] * (1.f / DIN) + EPSV);
    __syncthreads();
    xr[tid]          = d0 * rstd * ln_w[tid]          + ln_b[tid];
    xr[tid + NT]     = d1 * rstd * ln_w[tid + NT]     + ln_b[tid + NT];
    xr[tid + 2 * NT] = d2 * rstd * ln_w[tid + 2 * NT] + ln_b[tid + 2 * NT];
    __syncthreads();

    // ---- x_proj = ln(x) @ w_in ----
    if (tid < DD) {
      float a = 0.f;
      #pragma unroll 4
      for (int e = 0; e < DIN; ++e) a = fmaf(xr[e], w_in[e * DD + tid], a);
      s_xp[tid] = a;
    }
    __syncthreads();
    // ---- q = x_proj @ w_q (tick-invariant) ----
    if (tid < DD) {
      float a = 0.f;
      #pragma unroll 4
      for (int e = 0; e < DD; ++e) a = fmaf(s_xp[e], w_q[e * DD + tid], a);
      s_q[tid] = a;
    }
    __syncthreads();
    // ---- wkq = w_k @ q : scores_n = slots_n . wkq (kills the slots@w_k GEMM) ----
    if (tid < DD) {
      const float4* row = (const float4*)(w_k + tid * DD);
      float a = 0.f;
      #pragma unroll 4
      for (int i = 0; i < DD / 4; ++i) {
        float4 w4 = row[i];
        a = fmaf(w4.x, s_q[4 * i + 0], a);
        a = fmaf(w4.y, s_q[4 * i + 1], a);
        a = fmaf(w4.z, s_q[4 * i + 2], a);
        a = fmaf(w4.w, s_q[4 * i + 3], a);
      }
      s_wkq[tid] = a;
    }
    __syncthreads();

    for (int t = 0; t < MAXT; ++t) {
      // ---- attention scores: 4 threads per slot ----
      {
        int n = tid >> 2, q4 = tid & 3;
        const float* sl = s_slots + n * DD + q4 * 32;
        const float* wk = s_wkq + q4 * 32;
        float a = 0.f;
        #pragma unroll
        for (int i = 0; i < 32; ++i) a = fmaf(sl[i], wk[i], a);
        s_part[tid] = a;
      }
      __syncthreads();
      if (tid < NS) {
        float sc = (s_part[4 * tid] + s_part[4 * tid + 1] +
                    s_part[4 * tid + 2] + s_part[4 * tid + 3]) * rscale;
        s_attn[tid] = sc; s_red[tid] = sc;
      }
      __syncthreads();
      for (int off = NS / 2; off > 0; off >>= 1) {
        if (tid < off) s_red[tid] = fmaxf(s_red[tid], s_red[tid + off]);
        __syncthreads();
      }
      float smax = s_red[0];
      __syncthreads();
      if (tid < NS) { float e = __expf(s_attn[tid] - smax); s_attn[tid] = e; s_red[tid] = e; }
      __syncthreads();
      for (int off = NS / 2; off > 0; off >>= 1) {
        if (tid < off) s_red[tid] += s_red[tid + off];
        __syncthreads();
      }
      float sinv = 1.f / s_red[0];
      __syncthreads();
      if (tid < NS) s_attn[tid] *= sinv;
      __syncthreads();
      // ---- aslots = attn @ slots (kills the slots@w_v GEMM) ----
      if (tid < DD) {
        float a = 0.f;
        #pragma unroll 4
        for (int n = 0; n < NS; ++n) a = fmaf(s_attn[n], s_slots[n * DD + tid], a);
        s_asl[tid] = a;
      }
      __syncthreads();
      // ---- read_out = aslots @ w_v ----
      if (tid < DD) {
        float a = 0.f;
        #pragma unroll 4
        for (int e = 0; e < DD; ++e) a = fmaf(s_asl[e], w_v[e * DD + tid], a);
        s_read[tid] = a;
      }
      __syncthreads();
      // ---- h1 | gate | kvec in parallel across thread groups ----
      if (tid < 64) {
        float a = halt_b1[tid];
        #pragma unroll 4
        for (int e = 0; e < DD; ++e) a = fmaf(s_read[e], halt_w1[e * 64 + tid], a);
        s_h1[tid] = fmaxf(a, 0.f);
      } else if (tid < 128) {
        int n = tid - 64;
        float a = b_gate[n];
        #pragma unroll 4
        for (int e = 0; e < DD; ++e) a = fmaf(s_read[e], w_gate[e * 64 + n], a);
        s_gate[n] = sigf(a);
      } else {
        int d = tid - 128;
        float a = 0.f;
        #pragma unroll 4
        for (int e = 0; e < DD; ++e) a = fmaf(s_read[e], dn_wk[e * DD + d], a);
        s_kv[d] = a;
      }
      __syncthreads();
      // ---- vvec | beta | k-norms | halt scalar ----
      if (tid < DD) {
        float a = 0.f;
        #pragma unroll 4
        for (int e = 0; e < DD; ++e) a = fmaf(s_read[e], dn_wv[e * DD + tid], a);
        s_vv[tid] = a;
      } else if (tid < 128 + NH) {
        int h = tid - 128;
        float a = dn_bb[h];
        #pragma unroll 4
        for (int e = 0; e < DD; ++e) a = fmaf(s_read[e], dn_wb[e * NH + h], a);
        s_beta[h] = 2.f * sigf(a);
      } else if (tid < 128 + 2 * NH) {
        int h = tid - 128 - NH;
        float a = 0.f;
        #pragma unroll
        for (int i = 0; i < DHD; ++i) { float v = s_kv[h * DHD + i]; a = fmaf(v, v, a); }
        s_nrm[h] = sqrtf(a) + 1e-6f;
      } else if (tid == 128 + 2 * NH) {
        float a = halt_b2[0];
        #pragma unroll 4
        for (int j = 0; j < 64; ++j) a = fmaf(s_h1[j], halt_w2[j], a);
        s_scal[0] = sigf(a);
      }
      __syncthreads();
      if (tid < DD) s_kv[tid] = s_kv[tid] / s_nrm[tid >> 5];
      __syncthreads();
      // ---- kh[n][h] = k_h . slots[n,h,:] ----
      {
        int n = tid >> 2, h = tid & 3;
        const float* sl = s_slots + n * DD + h * DHD;
        const float* kv = s_kv + h * DHD;
        float a = 0.f;
        #pragma unroll
        for (int i = 0; i < DHD; ++i) a = fmaf(sl[i], kv[i], a);
        s_kh[tid] = a;
      }
      __syncthreads();
      // ---- h_new -> bf16 A staging: adjacent-d pairs => ds_load_b64 + packed ds_store_b32
      #pragma unroll
      for (int i = 0; i < (NS * DD) / (2 * NT); ++i) {
        int base = (tid + i * NT) * 2;          // even => d pair shares the same head
        int n = base >> 7, d0 = base & (DD - 1), h = d0 >> 5;
        float2 hv = *(const float2*)(s_slots + base);
        float beta = s_beta[h], kh = s_kh[n * NH + h];
        float v0 = s_lam[d0]     * hv.x + beta * (s_vv[d0]     - kh * s_kv[d0]);
        float v1 = s_lam[d0 + 1] * hv.y + beta * (s_vv[d0 + 1] - kh * s_kv[d0 + 1]);
        bf16x2 p = { (__bf16)v0, (__bf16)v1 };
        *(bf16x2*)(s_abf + base) = p;
      }
      __syncthreads();
      // ---- upd = h_new @ dn_wo via v_wmma_f32_16x16x32_bf16, fused gated update ----
      {
        const int rA = lane & 15;
        const int kA = (lane >> 4) << 4;    // A comps 0-15 K-base: 0 or 16
        #pragma unroll
        for (int mt = 0; mt < 4; ++mt) {
          f32x8 c = {0.f, 0.f, 0.f, 0.f, 0.f, 0.f, 0.f, 0.f};
          #pragma unroll
          for (int kt = 0; kt < 4; ++kt) {
            const __bf16* ap = s_abf + (mt * 16 + rA) * DD + kt * 32 + kA;
            bf16x8 lo = *(const bf16x8*)ap;         // K = base .. base+7
            bf16x8 hi = *(const bf16x8*)(ap + 8);   // K = base+8 .. base+15
            bf16x16 av = __builtin_shufflevector(lo, hi,
                0, 1, 2, 3, 4, 5, 6, 7, 8, 9, 10, 11, 12, 13, 14, 15);
            c = __builtin_amdgcn_wmma_f32_16x16x32_bf16(
                    false, av, false, bfr[kt], (short)0, c, false, false);
          }
          const int colg = wv * 16 + (lane & 15);
          const int rb   = mt * 16 + ((lane >> 4) << 3);
          #pragma unroll
          for (int r = 0; r < 8; ++r) {
            int row = rb + r;
            float g = s_gate[row];
            int idx = row * DD + colg;
            s_slots[idx] = fmaf(g, c[r] - s_slots[idx], s_slots[idx]);
          }
        }
      }
      __syncthreads();
      // ---- acc/rem update + global mean(halt) across the 8 batch workgroups ----
      float remv = s_scal[1];
      float halt = s_scal[0];
      if (tid < DD) s_acc[tid] = fmaf(remv, s_read[tid], s_acc[tid]);
      __syncthreads();
      if (tid == 0) {
        if (t < MAXT - 1) s_scal[1] = remv * (1.f - halt);
        if (t >= 1 && t <= MAXT - 2) {
          int cell = s * MAXT + t;
          atomicAdd(&ws_sum[cell], halt);
          __threadfence();
          atomicAdd(&ws_cnt[cell], 1u);
          unsigned cdone;
          do {
            cdone = __hip_atomic_load(&ws_cnt[cell], __ATOMIC_ACQUIRE,
                                      __HIP_MEMORY_SCOPE_AGENT);
            if (cdone < BB) __builtin_amdgcn_s_sleep(1);
          } while (cdone < BB);
          float tot = __hip_atomic_load(&ws_sum[cell], __ATOMIC_RELAXED,
                                        __HIP_MEMORY_SCOPE_AGENT);
          s_scal[2] = tot * (1.f / BB);
        } else {
          s_scal[2] = 0.f;
        }
      }
      __syncthreads();
      if (t >= 1 && s_scal[2] > THRV) break;   // uniform across block & grid
    }

    // ---- out[b][s] = acc @ w_out ----
    __syncthreads();
    #pragma unroll
    for (int i = 0; i < DIN / NT; ++i) {
      int o = tid + i * NT;
      float a = 0.f;
      #pragma unroll 4
      for (int d = 0; d < DD; ++d) a = fmaf(s_acc[d], w_out[d * DIN + o], a);
      out[((size_t)b * SS + s) * DIN + o] = a;
    }
    __syncthreads();
  }

  // ---- slots_final (vectorized b128) ----
  {
    float4* dst = (float4*)(out + (size_t)BB * SS * DIN + (size_t)b * NS * DD);
    #pragma unroll
    for (int i = 0; i < (NS * DD) / (4 * NT); ++i)
      dst[tid + i * NT] = ((const float4*)s_slots)[tid + i * NT];
  }
}

extern "C" void kernel_launch(void* const* d_in, const int* in_sizes, int n_in,
                              void* d_out, int out_size, void* d_ws, size_t ws_size,
                              hipStream_t stream) {
  (void)in_sizes; (void)n_in; (void)out_size; (void)ws_size;
  // zero the cross-workgroup halt-sync cells (S*MAXT float sums + uint counters)
  hipMemsetAsync(d_ws, 0, 8192, stream);

  const float* xin      = (const float*)d_in[0];
  const float* slot_mem = (const float*)d_in[1];
  const float* ln_w     = (const float*)d_in[2];
  const float* ln_b     = (const float*)d_in[3];
  const float* w_in     = (const float*)d_in[4];
  const float* w_q      = (const float*)d_in[5];
  const float* w_k      = (const float*)d_in[6];
  const float* w_v      = (const float*)d_in[7];
  const float* w_gate   = (const float*)d_in[8];
  const float* b_gate   = (const float*)d_in[9];
  const float* halt_w1  = (const float*)d_in[10];
  const float* halt_b1  = (const float*)d_in[11];
  const float* halt_w2  = (const float*)d_in[12];
  const float* halt_b2  = (const float*)d_in[13];
  const float* w_out    = (const float*)d_in[14];
  const float* dn_wk    = (const float*)d_in[15];
  const float* dn_wv    = (const float*)d_in[16];
  const float* dn_wb    = (const float*)d_in[17];
  const float* dn_bb    = (const float*)d_in[18];
  const float* dn_lam   = (const float*)d_in[19];
  const float* dn_wo    = (const float*)d_in[20];

  float*    ws_sum = (float*)d_ws;
  unsigned* ws_cnt = (unsigned*)((char*)d_ws + 4096);

  slotbank_kernel<<<dim3(BB), dim3(NT), 0, stream>>>(
      xin, slot_mem, ln_w, ln_b, w_in, w_q, w_k, w_v, w_gate, b_gate,
      halt_w1, halt_b1, halt_w2, halt_b2, w_out, dn_wk, dn_wv, dn_wb,
      dn_bb, dn_lam, dn_wo, (float*)d_out, ws_sum, ws_cnt);
}
